// CausalSelfAttention_7765300871362
// MI455X (gfx1250) — matrix operand order repaired and checked by reading, not compiled, not run on hardware
//
#include <hip/hip_runtime.h>
#include <cstdint>

// ---------------------------------------------------------------------------
// Types for CDNA5 WMMA (wave32, v_wmma_f32_16x16x32_bf16)
// ---------------------------------------------------------------------------
typedef __bf16 bf16_t;
typedef __attribute__((ext_vector_type(16))) __bf16 v16bf;
typedef __attribute__((ext_vector_type(8)))  __bf16 v8bf;
typedef __attribute__((ext_vector_type(8)))  float  v8f;

union FragAB {
    v16bf v;
    v8bf  h[2];
};

__device__ __forceinline__ v8f wmma_bf16(v16bf a, v16bf b, v8f c) {
    // D(f32 16x16) = A(bf16 16x32) * B(bf16 32x16) + C
    return __builtin_amdgcn_wmma_f32_16x16x32_bf16(
        /*neg_a=*/false, a, /*neg_b=*/false, b,
        /*c_mod=*/(short)0, c, /*reuse_a=*/false, /*reuse_b=*/false);
}

// Async global->LDS copy, 16B per lane (CDNA5 GLOBAL_LOAD_ASYNC_TO_LDS_B128,
// tracked by ASYNCcnt). Generic pointers to __shared__ hold the LDS byte
// offset in their low 32 bits (flat aperture truncation rule).
__device__ __forceinline__ void async_ld_b128(const void* gptr, void* lptr) {
    unsigned lds = (unsigned)(uintptr_t)lptr;
    asm volatile("global_load_async_to_lds_b128 %0, %1, off"
                 :: "v"(lds), "v"(gptr) : "memory");
}
// Wait until at most N async ops outstanding (in-order completion lets the
// newest 4 (next tile's DMA) remain in flight while the current tile is safe).
__device__ __forceinline__ void wait_async0() {
    asm volatile("s_wait_asynccnt 0x0" ::: "memory");
}
__device__ __forceinline__ void wait_async4() {
    asm volatile("s_wait_asynccnt 0x4" ::: "memory");
}

__device__ __forceinline__ float rmax16(float v) {
    v = fmaxf(v, __shfl_xor(v, 1));
    v = fmaxf(v, __shfl_xor(v, 2));
    v = fmaxf(v, __shfl_xor(v, 4));
    v = fmaxf(v, __shfl_xor(v, 8));
    return v;
}
__device__ __forceinline__ float rsum16(float v) {
    v += __shfl_xor(v, 1);
    v += __shfl_xor(v, 2);
    v += __shfl_xor(v, 4);
    v += __shfl_xor(v, 8);
    return v;
}

#define EMBED 1024
#define HEADS 16
#define HDIM  64
#define SEQ   2048
#define BATCH 2
#define MROWS (BATCH * SEQ)          // 4096
#define ATT_SCALE 0.125f             // 64^-0.5

// ---------------------------------------------------------------------------
// fp32 -> bf16 conversion
// ---------------------------------------------------------------------------
__global__ void f32_to_bf16_kernel(const float* __restrict__ in,
                                   bf16_t* __restrict__ out, int n) {
    int i = blockIdx.x * blockDim.x + threadIdx.x;
    if (i < n) out[i] = (bf16_t)in[i];
}

// ---------------------------------------------------------------------------
// Tiled bf16 GEMM:  C[m,n] = sum_k A[m,k] * Bw[n,k]  (+ bias[n])
//   A : [M,K] bf16 row-major      Bw : [N,K] bf16 row-major (torch Linear W)
// MODE 0: Cout fp32 [M,N]
// MODE 1: QKV split epilogue -> q[B,H,N,D], k[B,H,N,D], vT[B,H,D,N] (bf16)
// Block: 256 threads = 8 waves in 4(M) x 2(N); block tile 128x128;
// per-wave 32x64 = 2x4 WMMA tiles; K-step 32. Tiles are staged through the
// CDNA5 async global->LDS pipe with manual double buffering for overlap.
// ---------------------------------------------------------------------------
template <int MODE>
__global__ __launch_bounds__(256) void gemm_bf16_kernel(
    const bf16_t* __restrict__ A, const bf16_t* __restrict__ Bw,
    const float* __restrict__ bias, float* __restrict__ Cout,
    bf16_t* __restrict__ qp, bf16_t* __restrict__ kp, bf16_t* __restrict__ vtp,
    int M, int N, int K) {
    __shared__ __align__(16) bf16_t As[2][128][32];
    __shared__ __align__(16) bf16_t Bs[2][128][32];

    const int tid  = threadIdx.x;
    const int lane = tid & 31;
    const int wave = tid >> 5;
    const int wm   = wave >> 1;      // 0..3
    const int wn   = wave & 1;       // 0..1
    const int half = lane >> 4;      // 0/1
    const int l16  = lane & 15;
    const int rowBase = blockIdx.y * 128;
    const int colBase = blockIdx.x * 128;

    v8f acc[2][4];
    const v8f zf = {0.f, 0.f, 0.f, 0.f, 0.f, 0.f, 0.f, 0.f};
    for (int i = 0; i < 2; ++i)
        for (int j = 0; j < 4; ++j) acc[i][j] = zf;

    // This thread's two 16-byte tile segments (fixed across k-steps)
    const int seg0 = tid * 2;
    const int r0   = seg0 >> 2;
    const int cs0  = (seg0 & 3) * 8;
    const int r1   = (seg0 + 1) >> 2;
    const int cs1  = ((seg0 + 1) & 3) * 8;

    // Prologue: DMA first k-slab into buffer 0
    async_ld_b128(&A[(size_t)(rowBase + r0) * K + cs0], &As[0][r0][cs0]);
    async_ld_b128(&Bw[(size_t)(colBase + r0) * K + cs0], &Bs[0][r0][cs0]);
    async_ld_b128(&A[(size_t)(rowBase + r1) * K + cs1], &As[0][r1][cs1]);
    async_ld_b128(&Bw[(size_t)(colBase + r1) * K + cs1], &Bs[0][r1][cs1]);

    const int steps = K >> 5;
    for (int kk = 0; kk < steps; ++kk) {
        const int cur = kk & 1;
        if (kk + 1 < steps) {
            const int k1 = (kk + 1) << 5;
            async_ld_b128(&A[(size_t)(rowBase + r0) * K + k1 + cs0], &As[cur ^ 1][r0][cs0]);
            async_ld_b128(&Bw[(size_t)(colBase + r0) * K + k1 + cs0], &Bs[cur ^ 1][r0][cs0]);
            async_ld_b128(&A[(size_t)(rowBase + r1) * K + k1 + cs1], &As[cur ^ 1][r1][cs1]);
            async_ld_b128(&Bw[(size_t)(colBase + r1) * K + k1 + cs1], &Bs[cur ^ 1][r1][cs1]);
            wait_async4();           // current buffer complete; next 4 in flight
        } else {
            wait_async0();
        }
        __syncthreads();

        // A fragments: half0 -> K{0..7,16..23}, half1 -> K{8..15,24..31}
        FragAB af[2];
        for (int i = 0; i < 2; ++i) {
            int r = wm * 32 + i * 16 + l16;
            af[i].h[0] = *reinterpret_cast<const v8bf*>(&As[cur][r][half * 8]);
            af[i].h[1] = *reinterpret_cast<const v8bf*>(&As[cur][r][16 + half * 8]);
        }
        // B fragments: half0 -> K{0..7,16..23}, half1 -> K{8..15,24..31}
        FragAB bfr[4];
        for (int j = 0; j < 4; ++j) {
            int c = wn * 64 + j * 16 + l16;
            bfr[j].h[0] = *reinterpret_cast<const v8bf*>(&Bs[cur][c][half * 8]);
            bfr[j].h[1] = *reinterpret_cast<const v8bf*>(&Bs[cur][c][16 + half * 8]);
        }
        for (int i = 0; i < 2; ++i)
            for (int j = 0; j < 4; ++j)
                acc[i][j] = wmma_bf16(af[i].v, bfr[j].v, acc[i][j]);
        __syncthreads();
    }

    // Epilogue. C layout: elem e of v8f: row = e + 8*half, col = l16.
    for (int i = 0; i < 2; ++i) {
        for (int j = 0; j < 4; ++j) {
            int col = colBase + wn * 64 + j * 16 + l16;
            float bv = bias ? bias[col] : 0.f;
            if (MODE == 0) {
                for (int e = 0; e < 8; ++e) {
                    int row = rowBase + wm * 32 + i * 16 + e + half * 8;
                    Cout[(size_t)row * N + col] = acc[i][j][e] + bv;
                }
            } else {
                int part = col >> 10;        // 0=q 1=k 2=v
                int cc = col & (EMBED - 1);
                int hh = cc >> 6;
                int dd = cc & (HDIM - 1);
                for (int e = 0; e < 8; ++e) {
                    int row = rowBase + wm * 32 + i * 16 + e + half * 8;
                    int bb = row >> 11;          // / SEQ
                    int nn = row & (SEQ - 1);
                    bf16_t v = (bf16_t)(acc[i][j][e] + bv);
                    size_t headBase = (size_t)(bb * HEADS + hh);
                    if (part == 0)
                        qp[(headBase * SEQ + nn) * HDIM + dd] = v;
                    else if (part == 1)
                        kp[(headBase * SEQ + nn) * HDIM + dd] = v;
                    else
                        vtp[(headBase * HDIM + dd) * SEQ + nn] = v;
                }
            }
        }
    }
}

// ---------------------------------------------------------------------------
// Causal flash attention. Grid (SEQ/64, H, B); block = 128 threads (4 waves).
// Each wave owns 16 query rows; online softmax; 32 keys per step.
// K (32x64) and V^T (64x32) tiles are staged into LDS once per block via the
// async pipe (double buffered) and shared by all 4 waves.
//   q,k : [B,H,N,D] bf16   vT : [B,H,D,N] bf16   attn : [B*N, E] bf16
// ---------------------------------------------------------------------------
__global__ __launch_bounds__(128) void flash_attn_kernel(
    const bf16_t* __restrict__ q, const bf16_t* __restrict__ k,
    const bf16_t* __restrict__ vt, bf16_t* __restrict__ attn) {
    const int tid  = threadIdx.x;
    const int lane = tid & 31;
    const int wave = tid >> 5;
    const int half = lane >> 4;
    const int l16  = lane & 15;
    const int h    = blockIdx.y;
    const int b    = blockIdx.z;
    const int qb   = blockIdx.x * 64;
    const int qrow = qb + wave * 16;

    const bf16_t* qh = q  + (size_t)(b * HEADS + h) * SEQ * HDIM;
    const bf16_t* kh = k  + (size_t)(b * HEADS + h) * SEQ * HDIM;
    const bf16_t* vh = vt + (size_t)(b * HEADS + h) * HDIM * SEQ;

    __shared__ __align__(16) bf16_t Ks[2][32][64];   // [key][d]
    __shared__ __align__(16) bf16_t Vs[2][64][32];   // [d][key]
    __shared__ __align__(16) bf16_t pbuf[4][16][40]; // per-wave P staging

    // Per-thread DMA segment mapping (256 x 16B segs per tile, 2 per thread)
    // K tile: row = seg>>3 (0..31), col = (seg&7)*8
    // V tile: row = seg>>2 (0..63), col = (seg&3)*8
    const int segA = tid * 2, segB = tid * 2 + 1;
    auto stage_kv = [&](int kb, int buf) {
        int kra = kb + (segA >> 3), krb = kb + (segB >> 3);
        if (kra >= SEQ) kra = SEQ - 1;       // masked keys: clamp source
        if (krb >= SEQ) krb = SEQ - 1;
        async_ld_b128(&kh[(size_t)kra * HDIM + (segA & 7) * 8],
                      &Ks[buf][segA >> 3][(segA & 7) * 8]);
        async_ld_b128(&kh[(size_t)krb * HDIM + (segB & 7) * 8],
                      &Ks[buf][segB >> 3][(segB & 7) * 8]);
        int voa = kb + (segA & 3) * 8, vob = kb + (segB & 3) * 8;
        if (voa + 8 > SEQ) voa = SEQ - 8;
        if (vob + 8 > SEQ) vob = SEQ - 8;
        async_ld_b128(&vh[(size_t)(segA >> 2) * SEQ + voa],
                      &Vs[buf][segA >> 2][(segA & 3) * 8]);
        async_ld_b128(&vh[(size_t)(segB >> 2) * SEQ + vob],
                      &Vs[buf][segB >> 2][(segB & 3) * 8]);
    };

    // Load Q fragments for the wave's 16 rows (D split 0..31 / 32..63)
    FragAB aq[2];
    {
        size_t qr = (size_t)(qrow + l16) * HDIM;
        for (int c = 0; c < 2; ++c) {
            aq[c].h[0] = *reinterpret_cast<const v8bf*>(&qh[qr + c * 32 + half * 8]);
            aq[c].h[1] = *reinterpret_cast<const v8bf*>(&qh[qr + c * 32 + 16 + half * 8]);
        }
    }

    const v8f zf = {0.f, 0.f, 0.f, 0.f, 0.f, 0.f, 0.f, 0.f};
    v8f o[4] = {zf, zf, zf, zf};
    float m[8], l[8];
    for (int e = 0; e < 8; ++e) { m[e] = -1e30f; l[e] = 0.f; }

    // Block-uniform trip count (last wave's causal frontier); fully-masked
    // steps skip compute but keep barriers convergent.
    const int nkt = (qb + 64 + 31) >> 5;
    stage_kv(0, 0);
    for (int kk = 0; kk < nkt; ++kk) {
        const int kb  = kk * 32;
        const int cur = kk & 1;
        if (kk + 1 < nkt) {
            stage_kv(kb + 32, cur ^ 1);
            wait_async4();
        } else {
            wait_async0();
        }
        __syncthreads();

        if (kb < qrow + 16) {   // step has at least one unmasked key
            // ---- S = Q @ K^T (16 x 32) as two 16x16 tiles ----
            v8f s[2] = {zf, zf};
            for (int sub = 0; sub < 2; ++sub) {
                for (int c = 0; c < 2; ++c) {
                    FragAB bk;
                    bk.h[0] = *reinterpret_cast<const v8bf*>(
                        &Ks[cur][sub * 16 + l16][c * 32 + half * 8]);
                    bk.h[1] = *reinterpret_cast<const v8bf*>(
                        &Ks[cur][sub * 16 + l16][c * 32 + 16 + half * 8]);
                    s[sub] = wmma_bf16(aq[c].v, bk.v, s[sub]);
                }
            }

            // ---- scale + causal mask + online softmax ----
            float mt[8];
            for (int e = 0; e < 8; ++e) {
                int r  = qrow + e + half * 8;
                int c0 = kb + l16;
                int c1 = kb + 16 + l16;
                float a0 = (c0 <= r) ? s[0][e] * ATT_SCALE : -1e30f;
                float a1 = (c1 <= r) ? s[1][e] * ATT_SCALE : -1e30f;
                s[0][e] = a0;
                s[1][e] = a1;
                mt[e] = rmax16(fmaxf(a0, a1));
            }
            float alpha[8];
            for (int e = 0; e < 8; ++e) {
                float mn = fmaxf(m[e], mt[e]);
                alpha[e] = __expf(m[e] - mn);
                m[e] = mn;
            }
            for (int e = 0; e < 8; ++e) {
                float p0 = __expf(s[0][e] - m[e]);
                float p1 = __expf(s[1][e] - m[e]);
                s[0][e] = p0;
                s[1][e] = p1;
                l[e] = l[e] * alpha[e] + rsum16(p0 + p1);
            }
            for (int n = 0; n < 4; ++n)
                for (int e = 0; e < 8; ++e) o[n][e] *= alpha[e];

            // ---- re-layout P (C-frag) -> A-frag via per-wave LDS ----
            for (int e = 0; e < 8; ++e) {
                int r = e + half * 8;
                pbuf[wave][r][l16]      = (bf16_t)s[0][e];
                pbuf[wave][r][16 + l16] = (bf16_t)s[1][e];
            }
            __asm__ volatile("s_wait_dscnt 0x0" ::: "memory");
            __builtin_amdgcn_wave_barrier();
            FragAB pf;
            pf.h[0] = *reinterpret_cast<const v8bf*>(&pbuf[wave][l16][half * 8]);
            pf.h[1] = *reinterpret_cast<const v8bf*>(&pbuf[wave][l16][16 + half * 8]);
            __builtin_amdgcn_wave_barrier();

            // ---- O += P (16x32) @ V (32x64) ----
            for (int n = 0; n < 4; ++n) {
                FragAB vf;
                vf.h[0] = *reinterpret_cast<const v8bf*>(
                    &Vs[cur][n * 16 + l16][half * 8]);
                vf.h[1] = *reinterpret_cast<const v8bf*>(
                    &Vs[cur][n * 16 + l16][16 + half * 8]);
                o[n] = wmma_bf16(pf.v, vf.v, o[n]);
            }
        }
        __syncthreads();
    }

    // ---- normalize and store to attn [B*N, E] (head-interleaved) ----
    float rl[8];
    for (int e = 0; e < 8; ++e) rl[e] = 1.0f / l[e];
    for (int n = 0; n < 4; ++n) {
        int col = h * HDIM + n * 16 + l16;
        for (int e = 0; e < 8; ++e) {
            size_t row = (size_t)b * SEQ + qrow + e + half * 8;
            attn[row * EMBED + col] = (bf16_t)(o[n][e] * rl[e]);
        }
    }
}

// ---------------------------------------------------------------------------
// Host-side launch
// ---------------------------------------------------------------------------
extern "C" void kernel_launch(void* const* d_in, const int* in_sizes, int n_in,
                              void* d_out, int out_size, void* d_ws, size_t ws_size,
                              hipStream_t stream) {
    const float* x      = (const float*)d_in[0];   // [B,N,E]
    const float* qkv_w  = (const float*)d_in[1];   // [3E,E]
    const float* qkv_b  = (const float*)d_in[2];   // [3E]
    const float* out_w  = (const float*)d_in[3];   // [E,E]
    const float* out_b  = (const float*)d_in[4];   // [E]
    float* out = (float*)d_out;                    // [B,N,E] fp32

    char* ws = (char*)d_ws;
    size_t off = 0;
    auto alloc = [&](size_t bytes) -> void* {
        void* p = ws + off;
        off += (bytes + 255) & ~(size_t)255;
        return p;
    };
    const size_t nX  = (size_t)MROWS * EMBED;        // 4096*1024
    const size_t nW1 = (size_t)3 * EMBED * EMBED;    // 3072*1024
    const size_t nW2 = (size_t)EMBED * EMBED;

    bf16_t* x_bf   = (bf16_t*)alloc(nX * 2);
    bf16_t* w1_bf  = (bf16_t*)alloc(nW1 * 2);
    bf16_t* w2_bf  = (bf16_t*)alloc(nW2 * 2);
    bf16_t* q_bf   = (bf16_t*)alloc(nX * 2);
    bf16_t* k_bf   = (bf16_t*)alloc(nX * 2);
    bf16_t* vt_bf  = (bf16_t*)alloc(nX * 2);
    bf16_t* att_bf = (bf16_t*)alloc(nX * 2);
    (void)ws_size; (void)in_sizes; (void)n_in; (void)out_size;

    // 1) precision downconvert (compute path is bf16 WMMA w/ f32 accumulate)
    f32_to_bf16_kernel<<<(int)((nX  + 255) / 256), 256, 0, stream>>>(x,     x_bf,  (int)nX);
    f32_to_bf16_kernel<<<(int)((nW1 + 255) / 256), 256, 0, stream>>>(qkv_w, w1_bf, (int)nW1);
    f32_to_bf16_kernel<<<(int)((nW2 + 255) / 256), 256, 0, stream>>>(out_w, w2_bf, (int)nW2);

    // 2) fused QKV projection + per-head split (V transposed)
    gemm_bf16_kernel<1><<<dim3(3 * EMBED / 128, MROWS / 128), 256, 0, stream>>>(
        x_bf, w1_bf, qkv_b, nullptr, q_bf, k_bf, vt_bf,
        MROWS, 3 * EMBED, EMBED);

    // 3) causal flash attention
    flash_attn_kernel<<<dim3(SEQ / 64, HEADS, BATCH), 128, 0, stream>>>(
        q_bf, k_bf, vt_bf, att_bf);

    // 4) output projection
    gemm_bf16_kernel<0><<<dim3(EMBED / 128, MROWS / 128), 256, 0, stream>>>(
        att_bf, w2_bf, out_b, out, nullptr, nullptr, nullptr,
        MROWS, EMBED, EMBED);
}
